// MLP_1520418423206
// MI455X (gfx1250) — matrix operand order repaired and checked by reading, not compiled, not run on hardware
//
#include <hip/hip_runtime.h>

typedef __attribute__((ext_vector_type(16))) _Float16     v16h;
typedef __attribute__((ext_vector_type(8)))  float        v8f;
typedef __attribute__((ext_vector_type(4)))  unsigned int v4u;
typedef __attribute__((ext_vector_type(2)))  _Float16     h2;

#define HD    4096
#define MDIM  14336
#define TOK   64
#define NT    32          // output columns per workgroup
#define KSTEP 32          // K per WMMA step
#define LSTR  40          // LDS row stride in halves (80B -> 16B aligned rows)

__constant__ float NF4_TAB[16] = {
    -1.0f, -0.6961928009986877f, -0.5250730514526367f, -0.39491748809814453f,
    -0.28444138169288635f, -0.18477343022823334f, -0.09105003625154495f, 0.0f,
    0.07958029955625534f, 0.16093020141124725f, 0.24611230194568634f,
    0.33791524171829224f, 0.44070982933044434f, 0.5626170039176941f,
    0.8333333134651184f, 1.0f };

// ---- gfx1250 async global->LDS copy (ASYNCcnt path, CDNA5 ISA) ----------
__device__ __forceinline__ unsigned lds_off(const void* p) {
    return (unsigned)(size_t)(__attribute__((address_space(3))) const void*)p;
}
__device__ __forceinline__ void async_copy_b128(unsigned dst_lds, const void* src) {
    asm volatile("global_load_async_to_lds_b128 %0, %1, off"
                 :: "v"(dst_lds), "v"(src) : "memory");
}
__device__ __forceinline__ void wait_async() {
    asm volatile("s_wait_asynccnt 0x0" ::: "memory");
}

// Assemble a v16h WMMA fragment from two 16-byte LDS chunks.
__device__ __forceinline__ v16h frag2(const _Float16* p0, const _Float16* p1) {
    union { v16h v; v4u u[2]; } r;
    r.u[0] = *(const v4u*)p0;
    r.u[1] = *(const v4u*)p1;
    return r.v;
}

// Dequant 8 int32 NF4 codes with a 256-entry packed-half2 LUT:
// 4 ds_load_b32 + 4 v_pk_mul_f16 instead of 8 scalar lookups + f32 cvt chain.
__device__ __forceinline__ v4u dequant8(v4u a, v4u b, h2 s2,
                                        const unsigned* lut2) {
    unsigned i0 = ((a[1] & 15u) << 4) | (a[0] & 15u);
    unsigned i1 = ((a[3] & 15u) << 4) | (a[2] & 15u);
    unsigned i2 = ((b[1] & 15u) << 4) | (b[0] & 15u);
    unsigned i3 = ((b[3] & 15u) << 4) | (b[2] & 15u);
    union PU { unsigned u; h2 hh; } p, r;
    v4u o;
    p.u = lut2[i0]; r.hh = p.hh * s2; o[0] = r.u;
    p.u = lut2[i1]; r.hh = p.hh * s2; o[1] = r.u;
    p.u = lut2[i2]; r.hh = p.hh * s2; o[2] = r.u;
    p.u = lut2[i3]; r.hh = p.hh * s2; o[3] = r.u;
    return o;
}

__device__ __forceinline__ void build_lut(unsigned* lut2, int t) {
    union { unsigned u; h2 hh; } e;
    e.hh[0] = (_Float16)NF4_TAB[t & 15];
    e.hh[1] = (_Float16)NF4_TAB[(t >> 4) & 15];
    lut2[t] = e.u;   // 256 threads build all 256 entries
}

// ---------------------------------------------------------------------------
// Kernel 1: g = x@Wg^T, u = x@Wu^T (NF4 dequant on the fly), h = silu(g)*u
// Grid: MDIM/NT = 448 blocks of 256 threads (8 wave32).
// Threads 0-127 stage the gate tile, 128-255 the up tile (8 codes each, b128).
// ---------------------------------------------------------------------------
__global__ __launch_bounds__(256)
void mlp_gateup_nf4(const _Float16* __restrict__ x,
                    const int*  __restrict__ gc, const float* __restrict__ gsc,
                    const int*  __restrict__ uc, const float* __restrict__ usc,
                    _Float16* __restrict__ hout) {
    __shared__ unsigned lut2[256];
    __shared__ _Float16 xs[TOK * LSTR];
    __shared__ _Float16 wg[NT * LSTR];
    __shared__ _Float16 wu[NT * LSTR];

    const int t  = threadIdx.x;
    const int n0 = blockIdx.x * NT;
    build_lut(lut2, t);

    const int lane = t & 31;
    const int wv   = t >> 5;        // wave 0..7
    const int tm   = wv & 3;        // token tile (16 rows each)
    const int tn   = wv >> 2;       // n tile 0..1
    const int hi   = lane >> 4;     // half-wave select
    const int ln   = lane & 15;

    // activation staging: 64 rows x 32 halves, one async b128 per thread
    const int xrow = t >> 2, xkof = (t & 3) * 8;
    const unsigned xs_dst = lds_off(&xs[xrow * LSTR + xkof]);

    // weight staging: half the block per matrix, 8 codes per thread
    const bool   low = (t < 128);
    const int    wrow = (t & 127) >> 2;       // 0..31
    const int    wkof = (t & 3) * 8;          // 0/8/16/24
    const int*   wc   = low ? gc  : uc;
    const float* wsc  = low ? gsc : usc;
    _Float16*    wdst = low ? wg  : wu;

    v8f accg = {}; v8f accu = {};

    for (int k0 = 0; k0 < HD; k0 += KSTEP) {
        __syncthreads();
        async_copy_b128(xs_dst, x + (size_t)xrow * HD + k0 + xkof);
        {
            const size_t base = (size_t)(n0 + wrow) * HD + (size_t)(k0 + wkof);
            const int    blk  = (n0 + wrow) * (HD / 64) + (k0 + wkof) / 64;
            const float  s    = wsc[blk];
            const h2     s2   = { (_Float16)s, (_Float16)s };
            v4u a = *(const v4u*)(wc + base);
            v4u b = *(const v4u*)(wc + base + 4);
            if (k0 + KSTEP < HD)                    // global_prefetch_b8
                __builtin_prefetch(wc + base + KSTEP, 0, 1);
            *(v4u*)(&wdst[wrow * LSTR + wkof]) = dequant8(a, b, s2, lut2);
        }
        wait_async();
        __syncthreads();

        // A fragment 16x32 fp16, contiguous k order used here for A and B (lane<16: K 0-15, lane>=16: K 16-31)
        const _Float16* ar = &xs[(tm * 16 + ln) * LSTR];
        v16h av = frag2(ar + hi * 16, ar + hi * 16 + 8);

        const _Float16* bg = &wg[(tn * 16 + ln) * LSTR + hi * 16];
        const _Float16* bu = &wu[(tn * 16 + ln) * LSTR + hi * 16];
        v16h fg = frag2(bg, bg + 8);
        v16h fu = frag2(bu, bu + 8);
        accg = __builtin_amdgcn_wmma_f32_16x16x32_f16(
            false, av, false, fg, (short)0, accg, false, false);
        accu = __builtin_amdgcn_wmma_f32_16x16x32_f16(
            false, av, false, fu, (short)0, accu, false, false);
    }

    // epilogue: h = silu(g) * u
    const int col = n0 + tn * 16 + ln;
#pragma unroll
    for (int r = 0; r < 8; ++r) {
        float g = accg[r];
        float u = accu[r];
        float s = g / (1.0f + __expf(-g)) * u;
        const int tok = tm * 16 + hi * 8 + r;   // C layout: VGPR r = M r / r+8
        hout[(size_t)tok * MDIM + col] = (_Float16)s;
    }
}

// ---------------------------------------------------------------------------
// Kernel 2: out = h @ Wd^T (NF4 dequant on the fly). Grid: HD/NT = 128 blocks.
// ---------------------------------------------------------------------------
__global__ __launch_bounds__(256)
void mlp_down_nf4(const _Float16* __restrict__ hin,
                  const int*  __restrict__ dc, const float* __restrict__ dsc,
                  _Float16* __restrict__ out) {
    __shared__ unsigned lut2[256];
    __shared__ _Float16 xs[TOK * LSTR];
    __shared__ _Float16 wd[NT * LSTR];

    const int t  = threadIdx.x;
    const int n0 = blockIdx.x * NT;
    build_lut(lut2, t);

    const int lane = t & 31;
    const int wv   = t >> 5;
    const int tm   = wv & 3;
    const int tn   = wv >> 2;
    const int hi   = lane >> 4;
    const int ln   = lane & 15;

    const int xrow = t >> 2, xkof = (t & 3) * 8;
    const unsigned xs_dst = lds_off(&xs[xrow * LSTR + xkof]);

    const int wrow = (t & 127) >> 2;
    const int wkof = (t & 3) * 8;

    v8f acc = {};

    for (int k0 = 0; k0 < MDIM; k0 += KSTEP) {
        __syncthreads();
        async_copy_b128(xs_dst, hin + (size_t)xrow * MDIM + k0 + xkof);
        if (t < 128) {
            const size_t base = (size_t)(n0 + wrow) * MDIM + (size_t)(k0 + wkof);
            const int    blk  = (n0 + wrow) * (MDIM / 64) + (k0 + wkof) / 64;
            const float  s    = dsc[blk];
            const h2     s2   = { (_Float16)s, (_Float16)s };
            v4u a = *(const v4u*)(dc + base);
            v4u b = *(const v4u*)(dc + base + 4);
            if (k0 + KSTEP < MDIM)
                __builtin_prefetch(dc + base + KSTEP, 0, 1);
            *(v4u*)(&wd[wrow * LSTR + wkof]) = dequant8(a, b, s2, lut2);
        }
        wait_async();
        __syncthreads();

        const _Float16* ar = &xs[(tm * 16 + ln) * LSTR];
        v16h av = frag2(ar + hi * 16, ar + hi * 16 + 8);
        const _Float16* bd = &wd[(tn * 16 + ln) * LSTR + hi * 16];
        v16h fd = frag2(bd, bd + 8);
        acc = __builtin_amdgcn_wmma_f32_16x16x32_f16(
            false, av, false, fd, (short)0, acc, false, false);
    }

    const int col = n0 + tn * 16 + ln;
#pragma unroll
    for (int r = 0; r < 8; ++r) {
        const int tok = tm * 16 + hi * 8 + r;
        out[(size_t)tok * HD + col] = (_Float16)acc[r];
    }
}

extern "C" void kernel_launch(void* const* d_in, const int* in_sizes, int n_in,
                              void* d_out, int out_size, void* d_ws, size_t ws_size,
                              hipStream_t stream) {
    (void)in_sizes; (void)n_in; (void)out_size; (void)ws_size;
    const _Float16* x = (const _Float16*)d_in[0];
    const int*   gc   = (const int*)  d_in[1];
    const float* gsc  = (const float*)d_in[2];
    const int*   uc   = (const int*)  d_in[3];
    const float* usc  = (const float*)d_in[4];
    const int*   dc   = (const int*)  d_in[5];
    const float* dsc  = (const float*)d_in[6];
    _Float16* hbuf = (_Float16*)d_ws;        // [TOK, MDIM] fp16 intermediate
    _Float16* out  = (_Float16*)d_out;       // [TOK, HD] fp16

    mlp_gateup_nf4<<<dim3(MDIM / NT), dim3(256), 0, stream>>>(
        x, gc, gsc, uc, usc, hbuf);
    mlp_down_nf4<<<dim3(HD / NT), dim3(256), 0, stream>>>(
        hbuf, dc, dsc, out);
}
